// static_tdegnn_67233418052359
// MI455X (gfx1250) — compile-verified, matrix operand order checked by reading, not run on hardware
//
#include <hip/hip_runtime.h>
#include <hip/hip_bf16.h>
#include <math.h>

#define NNODES 50000
#define NEDGES 800000
#define NH     128
#define NLAY   4
#define NOUTC  40
#define HSTEP  0.1f

typedef __attribute__((ext_vector_type(16))) _Float16 v16h;
typedef __attribute__((ext_vector_type(8)))  _Float16 h8v;
typedef __attribute__((ext_vector_type(4)))  _Float16 h4v;
typedef __attribute__((ext_vector_type(8)))  float    v8f;
typedef int v4i_async __attribute__((vector_size(16)));

static const size_t NE_ELEM = (size_t)NNODES * NH;   // 6,400,000

#if defined(__has_builtin)
#if __has_builtin(__builtin_amdgcn_global_load_async_to_lds_b128) && \
    __has_builtin(__builtin_amdgcn_s_wait_asynccnt)
#define USE_ASYNC_LDS 1
#endif
#endif
#ifndef USE_ASYNC_LDS
#define USE_ASYNC_LDS 0
#endif

__device__ __forceinline__ float warp_sum(float v) {
#pragma unroll
  for (int off = 16; off > 0; off >>= 1) v += __shfl_xor(v, off, 32);
  return v;
}
__device__ __forceinline__ float elu1(float x) { return x > 0.f ? x : (expf(x) - 1.f); }

// ---------------- graph preprocessing ----------------
__global__ void k_zero_f(float* p, size_t n) {
  size_t i = (size_t)blockIdx.x * blockDim.x + threadIdx.x;
  if (i < n) p[i] = 0.f;
}
__global__ void k_deg(const int* __restrict__ ei, float* __restrict__ deg) {
  int e = blockIdx.x * blockDim.x + threadIdx.x;
  if (e < NEDGES) {
    int r = ei[e], c = ei[NEDGES + e];
    if (r != c) atomicAdd(&deg[r], 1.f);
  }
}
__global__ void k_dinv(const float* __restrict__ deg, float* __restrict__ dinv) {
  int i = blockIdx.x * blockDim.x + threadIdx.x;
  if (i < NNODES) {
    float d = deg[i];
    dinv[i] = d > 0.f ? rsqrtf(fmaxf(d, 1.f)) : 0.f;
  }
}
__global__ void k_lapw(const int* __restrict__ ei, const float* __restrict__ dinv,
                       float* __restrict__ lw) {
  int e = blockIdx.x * blockDim.x + threadIdx.x;
  if (e < NEDGES) {
    int r = ei[e], c = ei[NEDGES + e];
    lw[e] = (r != c) ? -(dinv[r] * dinv[c]) : 0.f;
  }
}
__global__ void k_f2h(const float* __restrict__ a, _Float16* __restrict__ b, size_t n) {
  size_t i = (size_t)blockIdx.x * blockDim.x + threadIdx.x;
  if (i < n) b[i] = (_Float16)a[i];
}

// ---------------- weight packing: W[O,128] f32 -> WMMA-B fragment layout f16 ----------------
// fragment (ct,ks): 32 lanes x 16 f16.  lane<16 -> col o=ct*16+lane, K pattern {kb..kb+7, kb+16..kb+23}
// lane>=16 -> same col group, K pattern {kb+8..kb+15, kb+24..kb+31},  kb = ks*32.
__global__ void k_pack_w(const float* __restrict__ W, _Float16* __restrict__ Bp,
                         int O, int colTiles) {
  int t = blockIdx.x * blockDim.x + threadIdx.x;
  int total = colTiles * 4 * 32 * 2;
  if (t >= total) return;
  int c = t & 1;
  int lane = (t >> 1) & 31;
  int frag = t >> 6;           // frag = ct*4 + ks
  int ks = frag & 3, ct = frag >> 2;
  int o = ct * 16 + (lane & 15);
  int joff = (lane < 16) ? (c ? 16 : 0) : (c ? 24 : 8);
  int kbase = ks * 32 + joff;
  _Float16* dst = Bp + ((size_t)frag * 32 + lane) * 16 + c * 8;
#pragma unroll
  for (int e = 0; e < 8; ++e) {
    float v = (o < O) ? W[(size_t)o * NH + kbase + e] : 0.f;
    dst[e] = (_Float16)v;
  }
}

// ---------------- WMMA GEMM with fused epilogues ----------------
// C[n,o] = sum_k A[n,k] * W[o,k] + bias[o], then epilogue.
// epi: 0=none  1=BN+ELU  2=BN+ReLU  3=reaction dT epilogue
// Packed B staged in LDS once per block (async-to-LDS path when available);
// each wave computes TWO 16-row tiles so every B fragment feeds two WMMAs.
__global__ __launch_bounds__(256) void k_gemm(
    const _Float16* __restrict__ A, const _Float16* __restrict__ Bp,
    const float* __restrict__ bias,
    float* __restrict__ Cf, _Float16* __restrict__ Ch,
    int nrows, int colTiles, int ncols, int ldc, int epi,
    const float* __restrict__ g, const float* __restrict__ bb,
    const float* __restrict__ mn, const float* __restrict__ vv,
    const float* __restrict__ U, const float* __restrict__ D2,
    const float* __restrict__ T1) {
  __shared__ _Float16 sB[8 * 4 * 32 * 16];   // up to 32 fragments (32 KB)

  // cooperative stage of packed B into LDS
  int nCh = colTiles * 4 * 32 * 2;           // 16-byte chunks
  for (int c = threadIdx.x; c < nCh; c += 256) {
#if USE_ASYNC_LDS
    __builtin_amdgcn_global_load_async_to_lds_b128(
        (__attribute__((address_space(1))) v4i_async*)(Bp + (size_t)c * 8),
        (__attribute__((address_space(3))) v4i_async*)(sB + (size_t)c * 8),
        0, 0);
#else
    *reinterpret_cast<h8v*>(sB + (size_t)c * 8) =
        *reinterpret_cast<const h8v*>(Bp + (size_t)c * 8);
#endif
  }
#if USE_ASYNC_LDS
  __builtin_amdgcn_s_wait_asynccnt(0);
#endif
  __syncthreads();

  int lane = threadIdx.x & 31;
  int wave = threadIdx.x >> 5;
  int row0 = (blockIdx.x * 8 + wave) * 32;   // two 16-row tiles per wave
  if (row0 >= nrows) return;                 // wave-uniform
  int m = lane & 15;
  int half = lane >> 4;
  int arow0 = row0 + m;
  int arow1 = row0 + 16 + m;

  v16h a0[4], a1[4];
#pragma unroll
  for (int ks = 0; ks < 4; ++ks) {
    h8v lo = {}, hi = {};
    if (arow0 < nrows) {
      const _Float16* Ar = A + (size_t)arow0 * NH + ks * 32 + half * 8;
      lo = *reinterpret_cast<const h8v*>(Ar);
      hi = *reinterpret_cast<const h8v*>(Ar + 16);
    }
#pragma unroll
    for (int i = 0; i < 8; ++i) { a0[ks][i] = lo[i]; a0[ks][8 + i] = hi[i]; }
    h8v lo1 = {}, hi1 = {};
    if (arow1 < nrows) {
      const _Float16* Ar = A + (size_t)arow1 * NH + ks * 32 + half * 8;
      lo1 = *reinterpret_cast<const h8v*>(Ar);
      hi1 = *reinterpret_cast<const h8v*>(Ar + 16);
    }
#pragma unroll
    for (int i = 0; i < 8; ++i) { a1[ks][i] = lo1[i]; a1[ks][8 + i] = hi1[i]; }
  }

  for (int ct = 0; ct < colTiles; ++ct) {
    int ocol = ct * 16 + m;
    float bv = (bias != nullptr && ocol < ncols) ? bias[ocol] : 0.f;
    v8f acc0, acc1;
#pragma unroll
    for (int i = 0; i < 8; ++i) { acc0[i] = bv; acc1[i] = bv; }
#pragma unroll
    for (int ks = 0; ks < 4; ++ks) {
      const _Float16* bp = sB + (((size_t)(ct * 4 + ks)) * 32 + lane) * 16;
      h8v lo = *reinterpret_cast<const h8v*>(bp);
      h8v hi = *reinterpret_cast<const h8v*>(bp + 8);
      v16h b;
#pragma unroll
      for (int i = 0; i < 8; ++i) { b[i] = lo[i]; b[8 + i] = hi[i]; }
      acc0 = __builtin_amdgcn_wmma_f32_16x16x32_f16(false, a0[ks], false, b,
                                                    (short)0, acc0, false, false);
      acc1 = __builtin_amdgcn_wmma_f32_16x16x32_f16(false, a1[ks], false, b,
                                                    (short)0, acc1, false, false);
    }
    if (ocol >= ncols) continue;
    float bnScale = 0.f, bnShift = 0.f;
    if (epi != 0) {
      float s = rsqrtf(vv[ocol] + 1e-5f) * g[ocol];
      bnScale = s;
      bnShift = bb[ocol] - mn[ocol] * s;
    }
#pragma unroll
    for (int tile = 0; tile < 2; ++tile) {
#pragma unroll
      for (int r = 0; r < 8; ++r) {
        int rr = row0 + tile * 16 + half * 8 + r;
        if (rr >= nrows) continue;
        float x = (tile == 0) ? acc0[r] : acc1[r];
        float y;
        if (epi == 0) {
          y = x;
        } else if (epi == 1) {
          y = elu1(x * bnScale + bnShift);
        } else if (epi == 2) {
          float t = x * bnScale + bnShift;
          y = t > 0.f ? t : 0.f;
        } else {
          size_t idx = (size_t)rr * NH + ocol;
          float t1 = T1[idx];
          float d2 = fminf(fmaxf(D2[idx], -1.f), 1.f);
          float dT = x + U[idx] + t1 * d2;
          float pre = t1 + HSTEP * dT;
          y = elu1(pre * bnScale + bnShift);
        }
        if (Cf) Cf[(size_t)rr * ldc + ocol] = y;
        if (Ch) Ch[(size_t)rr * ldc + ocol] = (_Float16)y;
      }
    }
  }
}

// ---------------- misc elementwise ----------------
__global__ void k_replicate(const float* __restrict__ z, float* a1, float* a2, float* a3,
                            _Float16* h1, _Float16* h2, _Float16* h3, size_t n) {
  size_t i = (size_t)blockIdx.x * blockDim.x + threadIdx.x;
  if (i < n) {
    float v = z[i];
    _Float16 h = (_Float16)v;
    a1[i] = v; a2[i] = v; a3[i] = v;
    h1[i] = h; h2[i] = h; h3[i] = h;
  }
}
__global__ void k_setup_scal(const float* __restrict__ mha, float* scal) {
  if (threadIdx.x == 0 && blockIdx.x == 0) scal[36] = fmaxf(mha[0], 0.f);
}
__global__ void k_layer_pre(float* scal) {
  if (threadIdx.x < 16) scal[16 + threadIdx.x] = 0.f;
}

// ---------------- attention statistics ----------------
__global__ __launch_bounds__(256) void k_attn(const _Float16* __restrict__ qh,
                                              const _Float16* __restrict__ kh,
                                              const float* __restrict__ scal,
                                              float* __restrict__ CmAcc) {
  __shared__ float sCm[16];
  if (threadIdx.x < 16) sCm[threadIdx.x] = 0.f;
  __syncthreads();
  int lane = threadIdx.x & 31;
  int wave = threadIdx.x >> 5;
  int node = blockIdx.x * 8 + wave;
  if (node < NNODES) {
    const size_t slot = NE_ELEM;
    float q[4][4], k[4][4];
#pragma unroll
    for (int i = 0; i < 4; ++i) {
      h4v qv = *reinterpret_cast<const h4v*>(qh + (size_t)i * slot + (size_t)node * NH + lane * 4);
      h4v kv = *reinterpret_cast<const h4v*>(kh + (size_t)i * slot + (size_t)node * NH + lane * 4);
#pragma unroll
      for (int e = 0; e < 4; ++e) { q[i][e] = (float)qv[e]; k[i][e] = (float)kv[e]; }
    }
    float s[16];
#pragma unroll
    for (int i = 0; i < 4; ++i)
#pragma unroll
      for (int j = 0; j < 4; ++j) {
        float acc = 0.f;
#pragma unroll
        for (int e = 0; e < 4; ++e) acc += q[i][e] * k[j][e];
        s[i * 4 + j] = acc;
      }
#pragma unroll
    for (int t = 0; t < 16; ++t) s[t] = warp_sum(s[t]);
    if (lane < 16) {
      int i = lane >> 2, j = lane & 3;
      const float scale = 0.08838834764831845f;   // 128^-0.5
      float r0 = s[i * 4 + 0] * scale, r1 = s[i * 4 + 1] * scale;
      float r2 = s[i * 4 + 2] * scale, r3 = s[i * 4 + 3] * scale;
      float mx = fmaxf(fmaxf(r0, r1), fmaxf(r2, r3));
      float e0 = expf(r0 - mx), e1 = expf(r1 - mx), e2 = expf(r2 - mx), e3 = expf(r3 - mx);
      float den = e0 + e1 + e2 + e3;
      float ej = (j == 0) ? e0 : (j == 1) ? e1 : (j == 2) ? e2 : e3;
      float attn = ej / den;
      atomicAdd(&sCm[lane], logf(attn * scal[36] + 1e-4f));
    }
  }
  __syncthreads();
  if (threadIdx.x < 16) atomicAdd(&CmAcc[threadIdx.x], sCm[threadIdx.x]);
}
__global__ void k_attn_fin(float* scal) {
  if (threadIdx.x == 0 && blockIdx.x == 0) {
    float c[4], ssum = 0.f;
#pragma unroll
    for (int j = 0; j < 4; ++j) {
      c[j] = 0.5f * (scal[16 + 12 + j] + scal[16 + j * 4 + 3]) / (float)NNODES;
      ssum += c[j];
    }
#pragma unroll
    for (int j = 0; j < 4; ++j) scal[32 + j] = c[j] / ssum;
  }
}
__global__ void k_combine(const float* __restrict__ a0, const float* __restrict__ a1,
                          const float* __restrict__ a2, const float* __restrict__ a3,
                          const float* __restrict__ scal,
                          float* __restrict__ T1, _Float16* __restrict__ T1h, size_t n) {
  size_t i = (size_t)blockIdx.x * blockDim.x + threadIdx.x;
  if (i < n) {
    float v = scal[32] * a0[i] + scal[33] * a1[i] + scal[34] * a2[i] + scal[35] * a3[i];
    T1[i] = v;
    T1h[i] = (_Float16)v;
  }
}

// ---------------- CG solver ----------------
__global__ void k_cg_init_pre(float* scal) {
  if (threadIdx.x == 0 && blockIdx.x == 0) { scal[5] = 0.f; scal[6] = 0.f; }
}
__global__ void k_cg_init(const float* __restrict__ X, float* __restrict__ R,
                          float* __restrict__ P, float* scal, size_t n) {
  size_t i = (size_t)blockIdx.x * blockDim.x + threadIdx.x;
  int lane = threadIdx.x & 31;
  float v = 0.f;
  if (i < n) { v = X[i]; R[i] = v; P[i] = v; }
  float w = warp_sum(v * v);
  if (lane == 0) atomicAdd(&scal[5], w);
}
__global__ void k_cg_pre(float* __restrict__ tmp, float* scal, size_t n) {
  size_t i = (size_t)blockIdx.x * blockDim.x + threadIdx.x;
  if (i < n) tmp[i] = 0.f;
  if (blockIdx.x == 0 && threadIdx.x < 3) scal[threadIdx.x] = 0.f;
}
__global__ __launch_bounds__(256) void k_spmv(const int* __restrict__ ei,
                                              const float* __restrict__ lw,
                                              const float* __restrict__ P,
                                              float* __restrict__ tmp,
                                              const float* __restrict__ scal) {
  if (scal[6] != 0.f) return;
  size_t t = (size_t)blockIdx.x * blockDim.x + threadIdx.x;
  int e = (int)(t >> 5);
  int grp = (int)(t & 31);
  float w = lw[e];
  if (w == 0.f) return;
  int r = ei[e], c = ei[NEDGES + e];
  float4 pv = *reinterpret_cast<const float4*>(P + (size_t)c * NH + grp * 4);
  float* ts = tmp + (size_t)r * NH + grp * 4;
  atomicAdd(ts + 0, w * pv.x);
  atomicAdd(ts + 1, w * pv.y);
  atomicAdd(ts + 2, w * pv.z);
  atomicAdd(ts + 3, w * pv.w);
}
__global__ void k_cg_lp(const float* __restrict__ P, const float* __restrict__ R,
                        const float* __restrict__ tmp, const float* __restrict__ Kap,
                        float* __restrict__ LP, float* scal, size_t n) {
  size_t i = (size_t)blockIdx.x * blockDim.x + threadIdx.x;
  int lane = threadIdx.x & 31;
  float rr = 0.f, plp = 0.f;
  if (scal[6] == 0.f && i < n) {
    int ch = (int)(i & (NH - 1));
    float kd = fminf(fmaxf(Kap[ch], 0.f), 1.f);
    float p = P[i];
    float lp = p + HSTEP * kd * (p + tmp[i]);
    LP[i] = lp;
    float r = R[i];
    rr = r * r;
    plp = p * lp;
  }
  rr = warp_sum(rr);
  plp = warp_sum(plp);
  if (lane == 0) { atomicAdd(&scal[0], rr); atomicAdd(&scal[1], plp); }
}
__global__ void k_cg_alpha(float* scal) {
  if (threadIdx.x == 0 && blockIdx.x == 0) scal[3] = scal[0] / (scal[1] + 1e-6f);
}
__global__ void k_cg_xr(float* __restrict__ X, float* __restrict__ R,
                        const float* __restrict__ P, const float* __restrict__ LP,
                        float* scal, size_t n) {
  size_t i = (size_t)blockIdx.x * blockDim.x + threadIdx.x;
  int lane = threadIdx.x & 31;
  float nr = 0.f;
  if (scal[6] == 0.f && i < n) {
    float a = scal[3];
    X[i] += a * P[i];
    float rn = R[i] - a * LP[i];
    R[i] = rn;
    nr = rn * rn;
  }
  nr = warp_sum(nr);
  if (lane == 0) atomicAdd(&scal[2], nr);
}
__global__ void k_cg_beta(float* scal) {
  if (threadIdx.x == 0 && blockIdx.x == 0) {
    if (scal[6] == 0.f) {
      float nr = scal[2];
      scal[4] = nr / (scal[5] + 1e-6f);
      if (nr < 1e-5f) scal[6] = 1.f; else scal[5] = nr;
    }
  }
}
__global__ void k_cg_p(float* __restrict__ P, const float* __restrict__ R,
                       const float* __restrict__ scal, size_t n) {
  size_t i = (size_t)blockIdx.x * blockDim.x + threadIdx.x;
  if (scal[6] == 0.f && i < n) P[i] = R[i] + scal[4] * P[i];
}
__global__ void k_store_act(const float* __restrict__ X, float* __restrict__ af,
                            _Float16* __restrict__ ah, size_t n) {
  size_t i = (size_t)blockIdx.x * blockDim.x + threadIdx.x;
  if (i < n) {
    float v = X[i];
    af[i] = v;
    ah[i] = (_Float16)v;
  }
}

// ---------------- host orchestration ----------------
extern "C" void kernel_launch(void* const* d_in, const int* in_sizes, int n_in,
                              void* d_out, int out_size, void* d_ws, size_t ws_size,
                              hipStream_t stream) {
  (void)in_sizes; (void)n_in; (void)out_size; (void)ws_size;
  const float* T       = (const float*)d_in[0];
  const int*   EI      = (const int*)d_in[1];
  const float* KopenW  = (const float*)d_in[2];
  const float* Kopenb  = (const float*)d_in[3];
  const float* bnOg = (const float*)d_in[4], *bnOb = (const float*)d_in[5];
  const float* bnOm = (const float*)d_in[6], *bnOv = (const float*)d_in[7];
  const float* convW   = (const float*)d_in[8];
  const float* convb   = (const float*)d_in[9];
  const float* bnAg = (const float*)d_in[10], *bnAb = (const float*)d_in[11];
  const float* bnAm = (const float*)d_in[12], *bnAv = (const float*)d_in[13];
  const float* Wq = (const float*)d_in[14], *bq = (const float*)d_in[15];
  const float* Wk = (const float*)d_in[16], *bk = (const float*)d_in[17];
  const float* mha = (const float*)d_in[18];
  const float* KR1W = (const float*)d_in[19], *KR1b = (const float*)d_in[20];
  const float* KR2W = (const float*)d_in[21], *KR2b = (const float*)d_in[22];
  const float* KRU0W = (const float*)d_in[23], *KRU0b = (const float*)d_in[24];
  const float* bng = (const float*)d_in[25], *bnb = (const float*)d_in[26];
  const float* bnm = (const float*)d_in[27], *bnv = (const float*)d_in[28];
  const float* Kappa = (const float*)d_in[29];
  const float* KcloseW = (const float*)d_in[30], *Kcloseb = (const float*)d_in[31];
  float* Out = (float*)d_out;

  // --- deterministic workspace carve ---
  char* wp = (char*)d_ws;
  auto carve = [&](size_t bytes) -> char* {
    char* r = wp;
    wp += (bytes + 255) & ~(size_t)255;
    return r;
  };
  const size_t NEf = NE_ELEM * sizeof(float);
  const size_t NEh = NE_ELEM * sizeof(_Float16);
  const size_t PK  = 8 * 4 * 32 * 16 * sizeof(_Float16);   // 128x128 pack
  const size_t PKc = 3 * 4 * 32 * 16 * sizeof(_Float16);   // Kclose pack

  float* deg   = (float*)carve(NNODES * sizeof(float));
  float* dinv  = (float*)carve(NNODES * sizeof(float));
  float* lapw  = (float*)carve(NEDGES * sizeof(float));
  float* scal  = (float*)carve(64 * sizeof(float));
  _Float16* pkOpen = (_Float16*)carve(PK);
  _Float16* pkConv = (_Float16*)carve(PK);
  _Float16* pkWq   = (_Float16*)carve(PK);
  _Float16* pkWk   = (_Float16*)carve(PK);
  _Float16* pkKR1[NLAY], *pkKR2[NLAY], *pkKRU0[NLAY];
  for (int j = 0; j < NLAY; ++j) pkKR1[j]  = (_Float16*)carve(PK);
  for (int j = 0; j < NLAY; ++j) pkKR2[j]  = (_Float16*)carve(PK);
  for (int j = 0; j < NLAY; ++j) pkKRU0[j] = (_Float16*)carve(PK);
  _Float16* pkClose = (_Float16*)carve(PKc);
  _Float16* Tin_h = (_Float16*)carve(NEh);
  float*    T0f   = (float*)carve(NEf);
  _Float16* T0h   = (_Float16*)carve(NEh);
  float*    actsF[NLAY]; _Float16* actsH[NLAY];
  for (int j = 0; j < NLAY; ++j) actsF[j] = (float*)carve(NEf);
  for (int j = 0; j < NLAY; ++j) actsH[j] = (_Float16*)carve(NEh);
  _Float16* qh = (_Float16*)carve(4 * NEh);
  _Float16* kh = (_Float16*)carve(4 * NEh);
  float*    T1f = (float*)carve(NEf);
  _Float16* T1h = (_Float16*)carve(NEh);
  float* Ubuf  = (float*)carve(NEf);
  float* D2buf = (float*)carve(NEf);
  float* Xb = (float*)carve(NEf);
  float* Rb = (float*)carve(NEf);
  float* Pb = (float*)carve(NEf);
  float* LPb = (float*)carve(NEf);
  float* tmpb = (float*)carve(NEf);

  const int EB = 256;
  const int gElem = (int)(NE_ELEM / EB);          // 25000, exact
  const int gEdge = NEDGES / EB;                  // 3125, exact
  const int gNode = (NNODES + EB - 1) / EB;
  const int gSpmv = (int)(((size_t)NEDGES * 32) / EB);  // 100000, exact
  const int gAttn = NNODES / 8;                   // 6250, exact
  const int gGemm = (NNODES + 255) / 256;         // 196: 8 waves x 32 rows per block
  const int gPack = (8 * 4 * 32 * 2 + EB - 1) / EB;
  const int gPackC = (3 * 4 * 32 * 2 + EB - 1) / EB;

  // graph preprocessing
  k_zero_f<<<gNode, EB, 0, stream>>>(deg, (size_t)NNODES);
  k_deg<<<gEdge, EB, 0, stream>>>(EI, deg);
  k_dinv<<<gNode, EB, 0, stream>>>(deg, dinv);
  k_lapw<<<gEdge, EB, 0, stream>>>(EI, dinv, lapw);
  k_setup_scal<<<1, 32, 0, stream>>>(mha, scal);

  // pack all weights into WMMA-B fragment layout
  k_pack_w<<<gPack, EB, 0, stream>>>(KopenW, pkOpen, NH, 8);
  k_pack_w<<<gPack, EB, 0, stream>>>(convW,  pkConv, NH, 8);
  k_pack_w<<<gPack, EB, 0, stream>>>(Wq,     pkWq,   NH, 8);
  k_pack_w<<<gPack, EB, 0, stream>>>(Wk,     pkWk,   NH, 8);
  for (int j = 0; j < NLAY; ++j) {
    k_pack_w<<<gPack, EB, 0, stream>>>(KR1W  + (size_t)j * NH * NH, pkKR1[j],  NH, 8);
    k_pack_w<<<gPack, EB, 0, stream>>>(KR2W  + (size_t)j * NH * NH, pkKR2[j],  NH, 8);
    k_pack_w<<<gPack, EB, 0, stream>>>(KRU0W + (size_t)j * NH * NH, pkKRU0[j], NH, 8);
  }
  k_pack_w<<<gPackC, EB, 0, stream>>>(KcloseW, pkClose, NOUTC, 3);

  // Th = elu(bn(T @ KopenW^T + b));  z = relu(bn(Th @ convW^T + b))
  k_f2h<<<gElem, EB, 0, stream>>>(T, Tin_h, NE_ELEM);
  k_gemm<<<gGemm, EB, 0, stream>>>(Tin_h, pkOpen, Kopenb, T0f, T0h,
                                   NNODES, 8, NH, NH, 1,
                                   bnOg, bnOb, bnOm, bnOv, nullptr, nullptr, nullptr);
  k_gemm<<<gGemm, EB, 0, stream>>>(T0h, pkConv, convb, actsF[0], actsH[0],
                                   NNODES, 8, NH, NH, 2,
                                   bnAg, bnAb, bnAm, bnAv, nullptr, nullptr, nullptr);
  k_replicate<<<gElem, EB, 0, stream>>>(actsF[0], actsF[1], actsF[2], actsF[3],
                                        actsH[1], actsH[2], actsH[3], NE_ELEM);

  int phys[NLAY] = {0, 1, 2, 3};
  int last = 0;
  for (int jj = 0; jj < NLAY; ++jj) {
    // attention over history
    k_layer_pre<<<1, 32, 0, stream>>>(scal);
    for (int i = 0; i < 4; ++i) {
      k_gemm<<<gGemm, EB, 0, stream>>>(actsH[phys[i]], pkWq, bq, nullptr,
                                       qh + (size_t)i * NE_ELEM,
                                       NNODES, 8, NH, NH, 0,
                                       nullptr, nullptr, nullptr, nullptr,
                                       nullptr, nullptr, nullptr);
      k_gemm<<<gGemm, EB, 0, stream>>>(actsH[phys[i]], pkWk, bk, nullptr,
                                       kh + (size_t)i * NE_ELEM,
                                       NNODES, 8, NH, NH, 0,
                                       nullptr, nullptr, nullptr, nullptr,
                                       nullptr, nullptr, nullptr);
    }
    k_attn<<<gAttn, EB, 0, stream>>>(qh, kh, scal, scal + 16);
    k_attn_fin<<<1, 32, 0, stream>>>(scal);
    k_combine<<<gElem, EB, 0, stream>>>(actsF[phys[0]], actsF[phys[1]],
                                        actsF[phys[2]], actsF[phys[3]],
                                        scal, T1f, T1h, NE_ELEM);
    // reaction: U = T0@KRU0^T+b ; D2 = T1@KR2^T+b ; X = elu(bn(T1 + H*dT))
    k_gemm<<<gGemm, EB, 0, stream>>>(T0h, pkKRU0[jj], KRU0b + (size_t)jj * NH,
                                     Ubuf, nullptr, NNODES, 8, NH, NH, 0,
                                     nullptr, nullptr, nullptr, nullptr,
                                     nullptr, nullptr, nullptr);
    k_gemm<<<gGemm, EB, 0, stream>>>(T1h, pkKR2[jj], KR2b + (size_t)jj * NH,
                                     D2buf, nullptr, NNODES, 8, NH, NH, 0,
                                     nullptr, nullptr, nullptr, nullptr,
                                     nullptr, nullptr, nullptr);
    k_gemm<<<gGemm, EB, 0, stream>>>(T1h, pkKR1[jj], KR1b + (size_t)jj * NH,
                                     Xb, nullptr, NNODES, 8, NH, NH, 3,
                                     bng + (size_t)jj * NH, bnb + (size_t)jj * NH,
                                     bnm + (size_t)jj * NH, bnv + (size_t)jj * NH,
                                     Ubuf, D2buf, T1f);
    // CG: (I + h*Kd*L) X = Tr
    k_cg_init_pre<<<1, 32, 0, stream>>>(scal);
    k_cg_init<<<gElem, EB, 0, stream>>>(Xb, Rb, Pb, scal, NE_ELEM);
    for (int it = 0; it < 10; ++it) {
      k_cg_pre<<<gElem, EB, 0, stream>>>(tmpb, scal, NE_ELEM);
      k_spmv<<<gSpmv, EB, 0, stream>>>(EI, lapw, Pb, tmpb, scal);
      k_cg_lp<<<gElem, EB, 0, stream>>>(Pb, Rb, tmpb, Kappa + (size_t)jj * NH,
                                        LPb, scal, NE_ELEM);
      k_cg_alpha<<<1, 32, 0, stream>>>(scal);
      k_cg_xr<<<gElem, EB, 0, stream>>>(Xb, Rb, Pb, LPb, scal, NE_ELEM);
      k_cg_beta<<<1, 32, 0, stream>>>(scal);
      k_cg_p<<<gElem, EB, 0, stream>>>(Pb, Rb, scal, NE_ELEM);
    }
    // rotate history ring: oldest slot overwritten by Tcur
    int dst = phys[0];
    k_store_act<<<gElem, EB, 0, stream>>>(Xb, actsF[dst], actsH[dst], NE_ELEM);
    int p0 = phys[0];
    phys[0] = phys[1]; phys[1] = phys[2]; phys[2] = phys[3]; phys[3] = p0;
    last = dst;
  }

  // output: Tcur @ KcloseW^T + b  -> [50000, 40]
  k_gemm<<<gGemm, EB, 0, stream>>>(actsH[last], pkClose, Kcloseb, Out, nullptr,
                                   NNODES, 3, NOUTC, NOUTC, 0,
                                   nullptr, nullptr, nullptr, nullptr,
                                   nullptr, nullptr, nullptr);
}
